// SchNetBlock_31576599560335
// MI455X (gfx1250) — compile-verified, matrix operand order checked by reading, not run on hardware
//
#include <hip/hip_runtime.h>
#include <math.h>

#define N_NODES 50000
#define N_EDGES 800000
#define HID  128
#define FILT 128
#define NG   50
#define CUTOFF_R 10.0f
#define LOG2F_ 0.69314718056f

typedef __attribute__((ext_vector_type(16))) _Float16 v16h;
typedef __attribute__((ext_vector_type(8)))  float    v8f;
typedef int vsi4 __attribute__((vector_size(16)));   // matches builtin param

#if defined(__has_builtin)
#if __has_builtin(__builtin_amdgcn_global_load_async_to_lds_b128)
#define USE_ASYNC_LDS 1
#endif
#endif
#ifndef USE_ASYNC_LDS
#define USE_ASYNC_LDS 0
#endif

// 16-bit A-matrix 16x32 layout (ISA 7.12.2): lanes 0-15 hold K={0..7,16..23},
// lanes 16-31 hold K={8..15,24..31}; element j -> K offset:
__device__ __forceinline__ int kmap(int j, int half) {
  return ((j >> 3) << 4) + (j & 7) + (half << 3);
}

// ---------- packed-B loads: one 32B per-lane vector load per fragment -------
// Packed layout produced by k_pack: frag (nt*nkt + kt), then lane, then j.
__device__ __forceinline__ v16h load_b_pk(const _Float16* __restrict__ p,
                                          int nkt, int kt, int nt) {
  const int lane = threadIdx.x & 31;
  return *(const v16h*)(p + ((((nt * nkt) + kt) * 32 + lane) << 4));
}

// A fragment (16x32) from an f16 row-major LDS tile (contiguous 8-half runs,
// vectorizes to ds_load_b128).
__device__ __forceinline__ v16h load_a_lds(const _Float16* __restrict__ src,
                                           int lda, int k0) {
  const int lane = threadIdx.x & 31;
  const int half = lane >> 4;
  const int r = lane & 15;
  v16h a;
#pragma unroll
  for (int j = 0; j < 16; ++j) a[j] = src[r * lda + k0 + kmap(j, half)];
  return a;
}

// Fast shifted softplus: v_exp_f32 + v_log_f32 (f16 matrix path dominates any
// precision delta from the precise log1p sequence).
__device__ __forceinline__ float ssp(float x) {
  float sp = (x > 15.0f) ? x : __logf(1.0f + __expf(x));
  return sp - LOG2F_;
}

__device__ __forceinline__ v8f wmma_f16(v16h a, v16h b, v8f c) {
  return __builtin_amdgcn_wmma_f32_16x16x32_f16(false, a, false, b, (short)0, c,
                                                false, false);
}

// Async global->LDS stage of `n16` 16-byte chunks (block-cooperative).
__device__ __forceinline__ void stage_lds_b128(void* lds, const void* g,
                                               int n16) {
#if USE_ASYNC_LDS
  for (int i = threadIdx.x; i < n16; i += (int)blockDim.x) {
    __builtin_amdgcn_global_load_async_to_lds_b128(
        (vsi4*)((char*)g + 16 * (size_t)i),
        (vsi4*)((char*)lds + 16 * (size_t)i), 0, 0);
  }
#if __has_builtin(__builtin_amdgcn_s_wait_asynccnt)
  __builtin_amdgcn_s_wait_asynccnt(0);
#else
  asm volatile("s_wait_asynccnt 0" ::: "memory");
#endif
#else
  for (int i = threadIdx.x; i < n16; i += (int)blockDim.x)
    ((uint4*)lds)[i] = ((const uint4*)g)[i];
#endif
}

// ---------------- weight repack: f32 [K x 128] -> f16 B-fragment layout -----
__global__ void k_pack(const float* __restrict__ src, _Float16* __restrict__ dst,
                       int nkt, int kmax) {
  int t = blockIdx.x * blockDim.x + threadIdx.x;
  int total = nkt * 8 * 512;
  if (t >= total) return;
  int j = t & 15;
  int lane = (t >> 4) & 31;
  int frag = t >> 9;                 // nt*nkt + kt
  int kt = frag % nkt, nt = frag / nkt;
  int half = lane >> 4;
  int k = kt * 32 + kmap(j, half);
  int c = nt * 16 + (lane & 15);
  dst[t] = (k < kmax) ? (_Float16)src[k * 128 + c] : (_Float16)0.0f;
}

__global__ void k_zero(float* __restrict__ p, int n) {
  int i = blockIdx.x * blockDim.x + threadIdx.x;
  if (i < n) p[i] = 0.0f;
}

// x[N,128] = h @ lin1_w. One block (128 thr, 4 waves) = one 16-row tile; each
// wave owns TWO 16-col tiles -> two independent WMMA chains hide the f16-WMMA
// RAW hazard and halve A-fragment LDS reads.
__global__ void __launch_bounds__(128) k_lin1(const float* __restrict__ h,
                                              const _Float16* __restrict__ wp,
                                              float* __restrict__ x) {
  __shared__ _Float16 aT[16 * HID];
  const int row0 = blockIdx.x * 16;
  for (int t = threadIdx.x; t < 16 * HID; t += 128)
    aT[t] = (_Float16)h[(size_t)(row0 + (t >> 7)) * HID + (t & 127)];
  __syncthreads();

  const int wv = threadIdx.x >> 5;
  const int lane = threadIdx.x & 31, half = lane >> 4;
  const int nt0 = wv * 2, nt1 = nt0 + 1;
  const int c0 = nt0 * 16 + (lane & 15), c1 = c0 + 16;
  v8f acc0 = {}, acc1 = {};
#pragma unroll
  for (int ks = 0; ks < 4; ++ks) {
    v16h a = load_a_lds(aT, HID, ks * 32);
    acc0 = wmma_f16(a, load_b_pk(wp, 4, ks, nt0), acc0);
    acc1 = wmma_f16(a, load_b_pk(wp, 4, ks, nt1), acc1);
  }
#pragma unroll
  for (int i = 0; i < 8; ++i) {
    x[(size_t)(row0 + i + half * 8) * FILT + c0] = acc0[i];
    x[(size_t)(row0 + i + half * 8) * FILT + c1] = acc1[i];
  }
}

// Fused edge pipeline: distance -> Gaussian smearing (A built in registers)
// -> WMMA layer1 + ssp (f16 via LDS) -> WMMA layer2 -> cosine cutoff
// -> gather x[col] (L2-resident) -> f32 atomic scatter into agg[row].
// One wave = 16 edges; 8 waves / block; packed weights async-staged in LDS;
// N-tiles processed in pairs for dual independent WMMA chains.
__global__ void __launch_bounds__(256) k_edge(
    const float* __restrict__ pos, const int* __restrict__ ei,
    const _Float16* __restrict__ w1p, const _Float16* __restrict__ w2p,
    const float* __restrict__ b1, const float* __restrict__ b2,
    const float* __restrict__ x, float* __restrict__ agg) {
  __shared__ _Float16 w1s[2 * 8 * 512];   // 16 KB packed layer-1 weights
  __shared__ _Float16 w2s[4 * 8 * 512];   // 32 KB packed layer-2 weights
  __shared__ _Float16 y1[8][16 * FILT];   // 32 KB layer-1 activations
  __shared__ float s_cut[8][16];
  __shared__ float s_d[8][16];
  __shared__ int s_row[8][16], s_col[8][16];

  stage_lds_b128(w1s, w1p, (2 * 8 * 512 * 2) / 16);
  stage_lds_b128(w2s, w2p, (4 * 8 * 512 * 2) / 16);

  const int wv = threadIdx.x >> 5;
  const int lane = threadIdx.x & 31;
  const int half = lane >> 4;
  const int e0 = (blockIdx.x * 8 + wv) * 16;

  if (lane < 16) {
    int e = e0 + lane;
    int r = ei[e], c = ei[N_EDGES + e];
    float dx = pos[r * 3 + 0] - pos[c * 3 + 0];
    float dy = pos[r * 3 + 1] - pos[c * 3 + 1];
    float dz = pos[r * 3 + 2] - pos[c * 3 + 2];
    float d = sqrtf(dx * dx + dy * dy + dz * dz);
    s_row[wv][lane] = r;
    s_col[wv][lane] = c;
    s_d[wv][lane] = d;
    s_cut[wv][lane] = 0.5f * (__cosf(d * 0.31415926535f) + 1.0f);
  }
  __syncthreads();

  // A fragments: Gaussian smearing computed straight into the WMMA layout.
  const float delta = CUTOFF_R / (float)(NG - 1);
  const float coeff = -0.5f / (delta * delta);
  const float d = s_d[wv][lane & 15];
  v16h a0, a1;
#pragma unroll
  for (int j = 0; j < 16; ++j) {
    int k = kmap(j, half);
    float t0 = d - (float)k * delta;
    a0[j] = (_Float16)__expf(coeff * t0 * t0);     // k <= 31 < NG: always valid
    int k2 = k + 32;
    float t1 = d - (float)k2 * delta;
    a1[j] = (k2 < NG) ? (_Float16)__expf(coeff * t1 * t1) : (_Float16)0.0f;
  }

  // Layer 1: [16,64(pad)] @ [64,128] + b1, shifted softplus -> f16 LDS
#pragma unroll
  for (int nt = 0; nt < 8; nt += 2) {
    const int c0 = nt * 16 + (lane & 15), c1 = c0 + 16;
    const float bias0 = b1[c0], bias1 = b1[c1];
    v8f acc0, acc1;
#pragma unroll
    for (int i = 0; i < 8; ++i) { acc0[i] = bias0; acc1[i] = bias1; }
    acc0 = wmma_f16(a0, load_b_pk(w1s, 2, 0, nt), acc0);
    acc1 = wmma_f16(a0, load_b_pk(w1s, 2, 0, nt + 1), acc1);
    acc0 = wmma_f16(a1, load_b_pk(w1s, 2, 1, nt), acc0);
    acc1 = wmma_f16(a1, load_b_pk(w1s, 2, 1, nt + 1), acc1);
#pragma unroll
    for (int i = 0; i < 8; ++i) {
      y1[wv][(i + half * 8) * FILT + c0] = (_Float16)ssp(acc0[i]);
      y1[wv][(i + half * 8) * FILT + c1] = (_Float16)ssp(acc1[i]);
    }
  }
  __syncthreads();

  // Hoist per-row edge metadata + 64-bit base pointers out of the hot loop.
  float mycut[8];
  const float* xb[8];
  float* ab[8];
#pragma unroll
  for (int i = 0; i < 8; ++i) {
    const int M = i + half * 8;
    mycut[i] = s_cut[wv][M];
    xb[i] = x + (size_t)s_col[wv][M] * FILT;
    ab[i] = agg + (size_t)s_row[wv][M] * FILT;
  }

  // Layer 2: [16,128] @ [128,128] + b2, * cutoff, then modulate+scatter
#pragma unroll
  for (int nt = 0; nt < 8; nt += 2) {
    const int c0 = nt * 16 + (lane & 15), c1 = c0 + 16;
    const float bias0 = b2[c0], bias1 = b2[c1];
    v8f acc0, acc1;
#pragma unroll
    for (int i = 0; i < 8; ++i) { acc0[i] = bias0; acc1[i] = bias1; }
#pragma unroll
    for (int ks = 0; ks < 4; ++ks) {
      v16h a = load_a_lds(y1[wv], FILT, ks * 32);
      acc0 = wmma_f16(a, load_b_pk(w2s, 4, ks, nt), acc0);
      acc1 = wmma_f16(a, load_b_pk(w2s, 4, ks, nt + 1), acc1);
    }
#pragma unroll
    for (int i = 0; i < 8; ++i) {
      atomicAdd(ab[i] + c0, xb[i][c0] * (acc0[i] * mycut[i]));
      atomicAdd(ab[i] + c1, xb[i][c1] * (acc1[i] * mycut[i]));
    }
  }
}

// Tail: out = ssp(agg @ lin2_w + lin2_b) @ lin_w + lin_b, LDS intermediate.
// 128 threads: 4 waves x 2 column tiles each.
__global__ void __launch_bounds__(128) k_tail(
    const float* __restrict__ agg, const _Float16* __restrict__ w2p,
    const float* __restrict__ b2, const _Float16* __restrict__ wop,
    const float* __restrict__ bo, float* __restrict__ out) {
  __shared__ _Float16 t0[16 * FILT];
  __shared__ _Float16 t1[16 * HID];
  const int row0 = blockIdx.x * 16;
  for (int t = threadIdx.x; t < 16 * FILT; t += 128)
    t0[t] = (_Float16)agg[(size_t)(row0 + (t >> 7)) * FILT + (t & 127)];
  __syncthreads();

  const int wv = threadIdx.x >> 5;
  const int lane = threadIdx.x & 31, half = lane >> 4;
  const int nt0 = wv * 2, nt1 = nt0 + 1;
  const int c0 = nt0 * 16 + (lane & 15), c1 = c0 + 16;
  {
    const float bias0 = b2[c0], bias1 = b2[c1];
    v8f acc0, acc1;
#pragma unroll
    for (int i = 0; i < 8; ++i) { acc0[i] = bias0; acc1[i] = bias1; }
#pragma unroll
    for (int ks = 0; ks < 4; ++ks) {
      v16h a = load_a_lds(t0, FILT, ks * 32);
      acc0 = wmma_f16(a, load_b_pk(w2p, 4, ks, nt0), acc0);
      acc1 = wmma_f16(a, load_b_pk(w2p, 4, ks, nt1), acc1);
    }
#pragma unroll
    for (int i = 0; i < 8; ++i) {
      t1[(i + half * 8) * HID + c0] = (_Float16)ssp(acc0[i]);
      t1[(i + half * 8) * HID + c1] = (_Float16)ssp(acc1[i]);
    }
  }
  __syncthreads();
  {
    const float bias0 = bo[c0], bias1 = bo[c1];
    v8f acc0, acc1;
#pragma unroll
    for (int i = 0; i < 8; ++i) { acc0[i] = bias0; acc1[i] = bias1; }
#pragma unroll
    for (int ks = 0; ks < 4; ++ks) {
      v16h a = load_a_lds(t1, HID, ks * 32);
      acc0 = wmma_f16(a, load_b_pk(wop, 4, ks, nt0), acc0);
      acc1 = wmma_f16(a, load_b_pk(wop, 4, ks, nt1), acc1);
    }
#pragma unroll
    for (int i = 0; i < 8; ++i) {
      out[(size_t)(row0 + i + half * 8) * HID + c0] = acc0[i];
      out[(size_t)(row0 + i + half * 8) * HID + c1] = acc1[i];
    }
  }
}

extern "C" void kernel_launch(void* const* d_in, const int* in_sizes, int n_in,
                              void* d_out, int out_size, void* d_ws,
                              size_t ws_size, hipStream_t stream) {
  (void)in_sizes; (void)n_in; (void)out_size; (void)ws_size;
  const float* h      = (const float*)d_in[0];
  const float* pos    = (const float*)d_in[1];
  const int*   ei     = (const int*)d_in[2];
  const float* mlp_w1 = (const float*)d_in[3];
  const float* mlp_b1 = (const float*)d_in[4];
  const float* mlp_w2 = (const float*)d_in[5];
  const float* mlp_b2 = (const float*)d_in[6];
  const float* lin1_w = (const float*)d_in[7];
  const float* lin2_w = (const float*)d_in[8];
  const float* lin2_b = (const float*)d_in[9];
  const float* lin_w  = (const float*)d_in[10];
  const float* lin_b  = (const float*)d_in[11];
  float* out = (float*)d_out;

  const size_t NF = (size_t)N_NODES * FILT;
  float* x   = (float*)d_ws;            // [N, FILT] f32
  float* agg = x + NF;                  // [N, FILT] f32
  _Float16* w1p = (_Float16*)(agg + NF);        // 2*8*512 halves
  _Float16* w2p = w1p + 2 * 8 * 512;            // 4*8*512 halves
  _Float16* l1p = w2p + 4 * 8 * 512;
  _Float16* l2p = l1p + 4 * 8 * 512;
  _Float16* lop = l2p + 4 * 8 * 512;

  // Repack weights into f16 WMMA B-fragment layout (tiny, L2-resident).
  k_pack<<<(2 * 8 * 512 + 255) / 256, 256, 0, stream>>>(mlp_w1, w1p, 2, NG);
  k_pack<<<(4 * 8 * 512 + 255) / 256, 256, 0, stream>>>(mlp_w2, w2p, 4, FILT);
  k_pack<<<(4 * 8 * 512 + 255) / 256, 256, 0, stream>>>(lin1_w, l1p, 4, HID);
  k_pack<<<(4 * 8 * 512 + 255) / 256, 256, 0, stream>>>(lin2_w, l2p, 4, FILT);
  k_pack<<<(4 * 8 * 512 + 255) / 256, 256, 0, stream>>>(lin_w,  lop, 4, HID);

  const int nAgg = (int)NF;
  k_zero<<<(nAgg + 255) / 256, 256, 0, stream>>>(agg, nAgg);
  k_lin1<<<N_NODES / 16, 128, 0, stream>>>(h, l1p, x);
  k_edge<<<N_EDGES / 128, 256, 0, stream>>>(pos, ei, w1p, w2p, mlp_b1, mlp_b2,
                                            x, agg);
  k_tail<<<N_NODES / 16, 128, 0, stream>>>(agg, l2p, lin2_b, lop, lin_b, out);
}
